// DNABERT2Model_10849087390532
// MI455X (gfx1250) — matrix operand order repaired and checked by reading, not compiled, not run on hardware
//
#include <hip/hip_runtime.h>
#include <hip/hip_bf16.h>
#include <math.h>

// ---------------------------------------------------------------------------
// DNABERT2-style encoder forward for MI455X (gfx1250), wave32 + WMMA bf16.
// GEMMs: LDS-staged double-buffered tiles via GLOBAL_LOAD_ASYNC_TO_LDS_B128
// (ASYNCcnt), fragments via ds_load_b128, math via v_wmma_f32_16x16x32_bf16.
// ---------------------------------------------------------------------------

#define V_  4096
#define E_  768
#define H_  12
#define NL_ 4
#define FFN_ 2048
#define L_  2048
#define HD_ 64

typedef __attribute__((ext_vector_type(16))) __bf16 v16bf;
typedef __attribute__((ext_vector_type(8)))  float  v8f;
typedef int vint4 __attribute__((vector_size(16)));   // matches builtin param

#define LDS_AS __attribute__((address_space(3)))
#define GLB_AS __attribute__((address_space(1)))

union Frag {
    v16bf  v;
    uint4  u[2];
    __bf16 h[16];
};

__device__ __forceinline__ v8f wmma_bf16(v16bf a, v16bf b, v8f c) {
    // (neg_a, A, neg_b, B, c_mod, C, reuse_a, reuse_b)
    return __builtin_amdgcn_wmma_f32_16x16x32_bf16(false, a, false, b, (short)0, c,
                                                   false, false);
}

// --- CDNA5 async global->LDS copy (ASYNCcnt path), with safe fallback -------
__device__ __forceinline__ void async_copy_b128(const __bf16* g, __bf16* l) {
#if __has_builtin(__builtin_amdgcn_global_load_async_to_lds_b128)
    __builtin_amdgcn_global_load_async_to_lds_b128(
        (GLB_AS vint4*)g, (LDS_AS vint4*)l, 0, 0);
#else
    *(uint4*)l = *(const uint4*)g;      // synchronous fallback
#endif
}

__device__ __forceinline__ void async_wait0() {
#if __has_builtin(__builtin_amdgcn_s_wait_asynccnt)
    __builtin_amdgcn_s_wait_asynccnt(0);
#else
#if defined(__AMDGCN__)
    asm volatile("s_wait_asynccnt 0" ::: "memory");
#endif
#endif
}

// A-fragment (16x32 bf16, M x K): lane l holds row M = l&15;
// element e maps to K = (e&7) + ((e>>3)<<4) + 8*(l>>4).   [ISA 7.12.2]
__device__ __forceinline__ v16bf load_a_frag(const __bf16* A, int lda,
                                             int m0, int k0, int lane) {
    const int hf = (lane >> 4) & 1, row = lane & 15;
    const __bf16* p = A + (size_t)(m0 + row) * lda + k0 + hf * 8;
    Frag f;
    f.u[0] = *(const uint4*)(p);        // K = k0+8h .. +7
    f.u[1] = *(const uint4*)(p + 16);   // K = k0+16+8h .. +7
    return f.v;
}

// B-fragment (32x16 bf16, K x N) from row-major W[N,K]:
// lane l holds column N = l&15; element e maps to K = k0 + (e&7) + ((e>>3)<<4) + 8*(l>>4).
__device__ __forceinline__ v16bf load_b_frag(const __bf16* W, int ldw,
                                             int n0, int k0, int lane) {
    const int hf = (lane >> 4) & 1, col = lane & 15;
    const __bf16* p = W + (size_t)(n0 + col) * ldw + k0 + hf * 8;
    Frag f;
    f.u[0] = *(const uint4*)(p);
    f.u[1] = *(const uint4*)(p + 16);
    return f.v;
}

// LDS variants: tile rows padded to 40 elems (80B) -> row r starts on bank
// 20r mod 64 = {0,4,...,60}: 16 concurrent ds_load_b128 are conflict-free.
#define SROW 40
__device__ __forceinline__ v16bf lds_a_frag(const __bf16* S, int m0, int lane) {
    const int hf = (lane >> 4) & 1, row = lane & 15;
    const __bf16* p = S + (m0 + row) * SROW + hf * 8;
    Frag f;
    f.u[0] = *(const uint4*)(p);
    f.u[1] = *(const uint4*)(p + 16);
    return f.v;
}
__device__ __forceinline__ v16bf lds_b_frag(const __bf16* S, int n0, int lane) {
    const int hf = (lane >> 4) & 1, col = lane & 15;
    const __bf16* p = S + (n0 + col) * SROW + hf * 8;
    Frag f;
    f.u[0] = *(const uint4*)(p);
    f.u[1] = *(const uint4*)(p + 16);
    return f.v;
}

// ---------------------------------------------------------------------------
// GEMM: C[M,N] = A[M,K] (bf16) @ W[N,K]^T (bf16) + bias[N] (+ Res)
// block = 256 threads = 8 waves (2 x 4); block tile 128x128, K-step 32.
// Wave tile 64x32 -> 8 WMMA / wave / K-step. A,B slabs double-buffered in
// LDS via async global->LDS B128 copies. Requires M%128==0, N%128==0, K%32==0.
// ---------------------------------------------------------------------------
template<int RES, int WF32, int WBF>
__global__ __launch_bounds__(256) void gemm_bf16_kernel(
    const __bf16* __restrict__ A, const __bf16* __restrict__ W,
    const float*  __restrict__ bias, const float* __restrict__ Res,
    float* __restrict__ Cf, __bf16* __restrict__ Cb,
    int M, int N, int K, int lda, int ldw, int ldc)
{
    __shared__ __align__(16) __bf16 smA[2][128 * SROW];
    __shared__ __align__(16) __bf16 smB[2][128 * SROW];

    const int tid  = threadIdx.x;
    const int lane = tid & 31;
    const int wave = tid >> 5;
    const int gm0 = blockIdx.y * 128;
    const int gn0 = blockIdx.x * 128;
    const int wm  = (wave >> 2) * 64;   // wave row offset within block tile
    const int wn  = (wave & 3) * 32;    // wave col offset within block tile

    v8f acc[4][2];
#pragma unroll
    for (int i = 0; i < 4; ++i)
#pragma unroll
        for (int j = 0; j < 2; ++j)
#pragma unroll
            for (int r = 0; r < 8; ++r) acc[i][j][r] = 0.0f;

    // stage one 128x32 K-slab of A and W into LDS buffer `buf`
    auto stage = [&](int buf, int k0) {
#pragma unroll
        for (int i = 0; i < 2; ++i) {
            const int idx = tid * 2 + i;           // 0..511
            const int row = idx >> 2, ch = (idx & 3) * 8;   // 4 x b128 per row
            async_copy_b128(A + (size_t)(gm0 + row) * lda + k0 + ch,
                            &smA[buf][row * SROW + ch]);
            async_copy_b128(W + (size_t)(gn0 + row) * ldw + k0 + ch,
                            &smB[buf][row * SROW + ch]);
        }
    };

    const int nsteps = K >> 5;
    stage(0, 0);
    for (int s = 0; s < nsteps; ++s) {
        async_wait0();        // my slab writes landed
        __syncthreads();      // everyone's landed; prev buffer fully consumed
        if (s + 1 < nsteps) stage((s + 1) & 1, (s + 1) << 5);

        const __bf16* bA = &smA[s & 1][0];
        const __bf16* bB = &smB[s & 1][0];
        v16bf af[4], bf[2];
#pragma unroll
        for (int i = 0; i < 4; ++i) af[i] = lds_a_frag(bA, wm + i * 16, lane);
#pragma unroll
        for (int j = 0; j < 2; ++j) bf[j] = lds_b_frag(bB, wn + j * 16, lane);
#pragma unroll
        for (int i = 0; i < 4; ++i)
#pragma unroll
            for (int j = 0; j < 2; ++j)
                acc[i][j] = wmma_bf16(af[i], bf[j], acc[i][j]);
    }

    const int hf = lane >> 4, col = lane & 15;
#pragma unroll
    for (int j = 0; j < 2; ++j) {
        const int n = gn0 + wn + j * 16 + col;
        const float bj = bias ? bias[n] : 0.0f;
#pragma unroll
        for (int i = 0; i < 4; ++i) {
#pragma unroll
            for (int r = 0; r < 8; ++r) {
                const int m = gm0 + wm + i * 16 + hf * 8 + r;  // C/D layout
                float val = acc[i][j][r] + bj;
                if (RES)  val += Res[(size_t)m * ldc + n];
                if (WF32) Cf[(size_t)m * ldc + n] = val;
                if (WBF)  Cb[(size_t)m * ldc + n] = (__bf16)val;
            }
        }
    }
}

// ---------------------------------------------------------------------------
// Flash attention: one wave per (head, 16-query tile). WMMA for QK^T and PV,
// online softmax with fp32 state; ALiBi bias + float key-padding term.
// ---------------------------------------------------------------------------
__global__ __launch_bounds__(32) void attention_kernel(
    const __bf16* __restrict__ qkv,   // (L, 3E) bf16
    const __bf16* __restrict__ vt,    // (E, L)  bf16  (V transposed per head)
    const int*    __restrict__ amask, // (L,) int
    __bf16*       __restrict__ o)     // (L, E) bf16
{
    __shared__ __align__(16) __bf16 psm[16 * 32];
    const int lane = threadIdx.x;
    const int hf = lane >> 4, col = lane & 15;
    const int h  = blockIdx.y;
    const int q0 = blockIdx.x * 16;

    // ALiBi slopes for H=12: h<8 -> 2^-(h+1) ; h>=8 -> 2^-((h-8)+0.5)
    const float slope = (h < 8) ? exp2f(-(float)(h + 1))
                                : exp2f(-((float)(h - 8) + 0.5f));

    const __bf16* Qb  = qkv + h * HD_;
    const __bf16* Kb  = qkv + E_ + h * HD_;
    const __bf16* Vtb = vt + (size_t)h * HD_ * L_;

    const v16bf qa0 = load_a_frag(Qb, 3 * E_, q0, 0,  lane);
    const v16bf qa1 = load_a_frag(Qb, 3 * E_, q0, 32, lane);

    v8f   oacc[4];
    float mrun[8], lrun[8];
#pragma unroll
    for (int t = 0; t < 4; ++t)
#pragma unroll
        for (int r = 0; r < 8; ++r) oacc[t][r] = 0.0f;
#pragma unroll
    for (int r = 0; r < 8; ++r) { mrun[r] = -1e30f; lrun[r] = 0.0f; }

    for (int kb = 0; kb < L_; kb += 32) {
        v8f s0, s1;
#pragma unroll
        for (int r = 0; r < 8; ++r) { s0[r] = 0.0f; s1[r] = 0.0f; }
        v16bf k00 = load_b_frag(Kb, 3 * E_, kb,      0,  lane);
        v16bf k01 = load_b_frag(Kb, 3 * E_, kb,      32, lane);
        v16bf k10 = load_b_frag(Kb, 3 * E_, kb + 16, 0,  lane);
        v16bf k11 = load_b_frag(Kb, 3 * E_, kb + 16, 32, lane);
        s0 = wmma_bf16(qa0, k00, s0); s0 = wmma_bf16(qa1, k01, s0);
        s1 = wmma_bf16(qa0, k10, s1); s1 = wmma_bf16(qa1, k11, s1);

        const float pad0 = (amask[kb + col] == 0)      ? 1.0f : 0.0f;
        const float pad1 = (amask[kb + 16 + col] == 0) ? 1.0f : 0.0f;
        const float k0f = (float)(kb + col), k1f = (float)(kb + 16 + col);

#pragma unroll
        for (int r = 0; r < 8; ++r) {
            const float qi = (float)(q0 + hf * 8 + r);
            float v0 = s0[r] * 0.125f + slope * (k0f - qi) + pad0; // 1/sqrt(64)
            float v1 = s1[r] * 0.125f + slope * (k1f - qi) + pad1;
            float t = fmaxf(v0, v1);
#pragma unroll
            for (int off = 8; off >= 1; off >>= 1)
                t = fmaxf(t, __shfl_xor(t, off, 16));
            const float mnew  = fmaxf(mrun[r], t);
            const float alpha = __expf(mrun[r] - mnew);
            const float p0 = __expf(v0 - mnew), p1 = __expf(v1 - mnew);
            float rs = p0 + p1;
#pragma unroll
            for (int off = 8; off >= 1; off >>= 1)
                rs += __shfl_xor(rs, off, 16);
            lrun[r] = lrun[r] * alpha + rs;
            mrun[r] = mnew;
#pragma unroll
            for (int t4 = 0; t4 < 4; ++t4) oacc[t4][r] *= alpha;
            psm[(hf * 8 + r) * 32 + col]      = (__bf16)p0;   // C layout stash
            psm[(hf * 8 + r) * 32 + 16 + col] = (__bf16)p1;
        }
        __syncthreads();

        Frag pf;                                   // re-layout C -> A fragment
#pragma unroll
        for (int e = 0; e < 16; ++e)
            pf.h[e] = psm[col * 32 + (e & 7) + ((e >> 3) << 4) + hf * 8];
        __syncthreads();

#pragma unroll
        for (int t4 = 0; t4 < 4; ++t4) {
            v16bf vb = load_b_frag(Vtb, L_, t4 * 16, kb, lane);
            oacc[t4] = wmma_bf16(pf.v, vb, oacc[t4]);
        }
    }

#pragma unroll
    for (int t4 = 0; t4 < 4; ++t4)
#pragma unroll
        for (int r = 0; r < 8; ++r) {
            const int m = q0 + hf * 8 + r;
            o[(size_t)m * E_ + h * HD_ + t4 * 16 + col] =
                (__bf16)(oacc[t4][r] / lrun[r]);
        }
}

// ---------------------------------------------------------------------------
// Elementwise / utility kernels
// ---------------------------------------------------------------------------
__global__ void cvt_bf16_kernel(const float* __restrict__ in,
                                __bf16* __restrict__ out, size_t n) {
    size_t i = (size_t)blockIdx.x * blockDim.x + threadIdx.x;
    const size_t stride = (size_t)gridDim.x * blockDim.x;
    for (; i < n; i += stride) out[i] = (__bf16)in[i];
}

__global__ void embed_kernel(const int* __restrict__ tokens,
                             const float* __restrict__ embed,
                             float* __restrict__ x) {
    const size_t idx = (size_t)blockIdx.x * blockDim.x + threadIdx.x;
    if (idx >= (size_t)L_ * E_) return;
    const int l = (int)(idx / E_), e = (int)(idx % E_);
    x[idx] = embed[(size_t)tokens[l] * E_ + e];
}

__global__ __launch_bounds__(256) void layernorm_kernel(
    const float* __restrict__ x, const float* __restrict__ g,
    const float* __restrict__ bta, __bf16* __restrict__ out, int N)
{
    const int row = blockIdx.x;
    const float* xr = x + (size_t)row * N;
    __shared__ float red0[8], red1[8];
    float s = 0.0f, s2 = 0.0f;
    for (int j = threadIdx.x; j < N; j += blockDim.x) {
        const float v = xr[j]; s += v; s2 += v * v;
    }
#pragma unroll
    for (int off = 16; off >= 1; off >>= 1) {
        s  += __shfl_down(s,  off);
        s2 += __shfl_down(s2, off);
    }
    const int w = threadIdx.x >> 5, ln = threadIdx.x & 31;
    if (ln == 0) { red0[w] = s; red1[w] = s2; }
    __syncthreads();
    if (threadIdx.x == 0) {
        float a = 0.0f, b = 0.0f;
#pragma unroll
        for (int i = 0; i < 8; ++i) { a += red0[i]; b += red1[i]; }
        red0[0] = a; red1[0] = b;
    }
    __syncthreads();
    const float mean = red0[0] / (float)N;
    const float var  = red1[0] / (float)N - mean * mean;
    const float rstd = rsqrtf(var + 1e-5f);
    for (int j = threadIdx.x; j < N; j += blockDim.x)
        out[(size_t)row * N + j] = (__bf16)((xr[j] - mean) * rstd * g[j] + bta[j]);
}

__device__ __forceinline__ float gelu_exact(float v) {
    return 0.5f * v * (1.0f + erff(v * 0.70710678118654752f));
}

__global__ void geglu_kernel(const float* __restrict__ u, __bf16* __restrict__ out) {
    const size_t idx = (size_t)blockIdx.x * blockDim.x + threadIdx.x;
    if (idx >= (size_t)L_ * FFN_) return;
    const int m = (int)(idx / FFN_), j = (int)(idx % FFN_);
    const float a = u[(size_t)m * 2 * FFN_ + j];
    const float g = u[(size_t)m * 2 * FFN_ + FFN_ + j];
    out[idx] = (__bf16)(a * gelu_exact(g));
}

__global__ void gelu_kernel(float* __restrict__ t, size_t n) {
    const size_t idx = (size_t)blockIdx.x * blockDim.x + threadIdx.x;
    if (idx >= n) return;
    t[idx] = gelu_exact(t[idx]);
}

__global__ void vt_kernel(const __bf16* __restrict__ qkv, __bf16* __restrict__ vt) {
    const size_t idx = (size_t)blockIdx.x * blockDim.x + threadIdx.x; // over E*L
    if (idx >= (size_t)E_ * L_) return;
    const int eo = (int)(idx / L_), l = (int)(idx % L_);
    vt[idx] = qkv[(size_t)l * 3 * E_ + 2 * E_ + eo];
}

// ---------------------------------------------------------------------------
// Host orchestration
// ---------------------------------------------------------------------------
static inline char* bump(char*& p, size_t bytes) {
    char* r = p;
    p += (bytes + 255) & ~(size_t)255;
    return r;
}

extern "C" void kernel_launch(void* const* d_in, const int* in_sizes, int n_in,
                              void* d_out, int out_size, void* d_ws, size_t ws_size,
                              hipStream_t stream) {
    (void)in_sizes; (void)n_in; (void)out_size; (void)ws_size;
    const int*   tokens  = (const int*)d_in[0];
    const int*   amask   = (const int*)d_in[1];
    const float* embed   = (const float*)d_in[2];
    const float* norm1_g = (const float*)d_in[3];
    const float* norm1_b = (const float*)d_in[4];
    const float* in_w    = (const float*)d_in[5];
    const float* in_b    = (const float*)d_in[6];
    const float* out_w   = (const float*)d_in[7];
    const float* out_b   = (const float*)d_in[8];
    const float* ffn_g   = (const float*)d_in[9];
    const float* ffn_bt  = (const float*)d_in[10];
    const float* ffn_w1  = (const float*)d_in[11];
    const float* ffn_b1  = (const float*)d_in[12];
    const float* ffn_w2  = (const float*)d_in[13];
    const float* ffn_b2  = (const float*)d_in[14];
    const float* fin_g   = (const float*)d_in[15];
    const float* fin_b   = (const float*)d_in[16];
    const float* mlm_w1  = (const float*)d_in[17];
    const float* mlm_b1  = (const float*)d_in[18];
    const float* mlm_g   = (const float*)d_in[19];
    const float* mlm_bt  = (const float*)d_in[20];
    const float* mlm_w2  = (const float*)d_in[21];
    const float* mlm_b2  = (const float*)d_in[22];
    float* logits = (float*)d_out;

    char* wp = (char*)d_ws;
    __bf16* in_w_bf   = (__bf16*)bump(wp, (size_t)NL_ * 3 * E_ * E_ * 2);
    __bf16* out_w_bf  = (__bf16*)bump(wp, (size_t)NL_ * E_ * E_ * 2);
    __bf16* ffn_w1_bf = (__bf16*)bump(wp, (size_t)NL_ * 2 * FFN_ * E_ * 2);
    __bf16* ffn_w2_bf = (__bf16*)bump(wp, (size_t)NL_ * E_ * FFN_ * 2);
    __bf16* mlm_w1_bf = (__bf16*)bump(wp, (size_t)E_ * E_ * 2);
    __bf16* mlm_w2_bf = (__bf16*)bump(wp, (size_t)V_ * E_ * 2);
    float*  x    = (float*) bump(wp, (size_t)L_ * E_ * 4);
    __bf16* hbf  = (__bf16*)bump(wp, (size_t)L_ * FFN_ * 2);   // LN / GEGLU out
    __bf16* qkvb = (__bf16*)bump(wp, (size_t)L_ * 3 * E_ * 2);
    __bf16* vtbf = (__bf16*)bump(wp, (size_t)E_ * L_ * 2);
    __bf16* obf  = (__bf16*)bump(wp, (size_t)L_ * E_ * 2);
    float*  u    = (float*) bump(wp, (size_t)L_ * 2 * FFN_ * 4);
    float*  t1   = (float*) bump(wp, (size_t)L_ * E_ * 4);

    // --- weights -> bf16 (L2-resident working set) ---
    auto cvt = [&](const float* s, __bf16* d, size_t n) {
        cvt_bf16_kernel<<<2048, 256, 0, stream>>>(s, d, n);
    };
    cvt(in_w,   in_w_bf,   (size_t)NL_ * 3 * E_ * E_);
    cvt(out_w,  out_w_bf,  (size_t)NL_ * E_ * E_);
    cvt(ffn_w1, ffn_w1_bf, (size_t)NL_ * 2 * FFN_ * E_);
    cvt(ffn_w2, ffn_w2_bf, (size_t)NL_ * E_ * FFN_);
    cvt(mlm_w1, mlm_w1_bf, (size_t)E_ * E_);
    cvt(mlm_w2, mlm_w2_bf, (size_t)V_ * E_);

    // --- embedding gather ---
    embed_kernel<<<(L_ * E_) / 256, 256, 0, stream>>>(tokens, embed, x);

    const dim3 gE (E_ / 128,       L_ / 128);   // N=768
    const dim3 gQ (3 * E_ / 128,   L_ / 128);   // N=2304
    const dim3 gF1(2 * FFN_ / 128, L_ / 128);   // N=4096
    const dim3 gV (V_ / 128,       L_ / 128);   // N=4096

    for (int l = 0; l < NL_; ++l) {
        // pre-LN attention
        layernorm_kernel<<<L_, 256, 0, stream>>>(x, norm1_g + l * E_, norm1_b + l * E_, hbf, E_);
        gemm_bf16_kernel<0, 0, 1><<<gQ, 256, 0, stream>>>(
            hbf, in_w_bf + (size_t)l * 3 * E_ * E_, in_b + (size_t)l * 3 * E_,
            nullptr, nullptr, qkvb, L_, 3 * E_, E_, E_, E_, 3 * E_);
        vt_kernel<<<(E_ * L_) / 256, 256, 0, stream>>>(qkvb, vtbf);
        attention_kernel<<<dim3(L_ / 16, H_), 32, 0, stream>>>(qkvb, vtbf, amask, obf);
        gemm_bf16_kernel<1, 1, 0><<<gE, 256, 0, stream>>>(
            obf, out_w_bf + (size_t)l * E_ * E_, out_b + (size_t)l * E_,
            x, x, nullptr, L_, E_, E_, E_, E_, E_);
        // pre-LN GEGLU FFN
        layernorm_kernel<<<L_, 256, 0, stream>>>(x, ffn_g + l * E_, ffn_bt + l * E_, hbf, E_);
        gemm_bf16_kernel<0, 1, 0><<<gF1, 256, 0, stream>>>(
            hbf, ffn_w1_bf + (size_t)l * 2 * FFN_ * E_, ffn_b1 + (size_t)l * 2 * FFN_,
            nullptr, u, nullptr, L_, 2 * FFN_, E_, E_, E_, 2 * FFN_);
        geglu_kernel<<<(L_ * FFN_) / 256, 256, 0, stream>>>(u, hbf);
        gemm_bf16_kernel<1, 1, 0><<<gE, 256, 0, stream>>>(
            hbf, ffn_w2_bf + (size_t)l * E_ * FFN_, ffn_b2 + (size_t)l * E_,
            x, x, nullptr, L_, E_, FFN_, FFN_, FFN_, E_);
    }

    // MLM head
    layernorm_kernel<<<L_, 256, 0, stream>>>(x, fin_g, fin_b, hbf, E_);
    gemm_bf16_kernel<0, 1, 0><<<gE, 256, 0, stream>>>(
        hbf, mlm_w1_bf, mlm_b1, nullptr, t1, nullptr, L_, E_, E_, E_, E_, E_);
    gelu_kernel<<<(L_ * E_) / 256, 256, 0, stream>>>(t1, (size_t)L_ * E_);
    layernorm_kernel<<<L_, 256, 0, stream>>>(t1, mlm_g, mlm_bt, hbf, E_);
    gemm_bf16_kernel<0, 1, 0><<<gV, 256, 0, stream>>>(
        hbf, mlm_w2_bf, mlm_b2, nullptr, logits, nullptr, L_, V_, E_, E_, E_, V_);
}
